// Model_55886114455783
// MI455X (gfx1250) — matrix operand order repaired and checked by reading, not compiled, not run on hardware
//
#include <hip/hip_runtime.h>

// ---------------------------------------------------------------- types
typedef __attribute__((ext_vector_type(16))) _Float16 v16h;
typedef __attribute__((ext_vector_type(8)))  float    v8f;
typedef __attribute__((ext_vector_type(4)))  unsigned int u32x4;
typedef __attribute__((ext_vector_type(8)))  int      i32x8;
typedef __attribute__((ext_vector_type(4)))  int      i32x4;

// ---------------------------------------------------------------- sizes
#define B_    16
#define NCH   32
#define T_    512
#define E_    128
#define W_    17
#define F_    33
#define NFFT  64
#define HOP   32
#define TOPM  32
#define HID   512
#define PRED  96
#define L_    576          // (W-1)*HOP + NFFT
#define KTOT  (T_*E_)      // 65536
#define MBN   (B_*NCH)     // 512
#define KSTEP 64           // fc1 K-chunk per LDS stage
#define TWO_PI_64 0.09817477042f   // 2*pi/64

#if defined(__AMDGCN__) && __has_builtin(__builtin_amdgcn_tensor_load_to_lds)
#define HAVE_TDM 1
#warning CDNA5 probe: tensor_load_to_lds builtin ENABLED in this pass
#else
#define HAVE_TDM 0
#warning CDNA5 probe: tensor_load_to_lds builtin NOT available in this pass (fallback loads)
#endif

__device__ __forceinline__ float hannw(int j) {
    return 0.5f - 0.5f * __cosf(TWO_PI_64 * (float)j);
}

__device__ __forceinline__ v8f wmma16(v16h a, v16h b, v8f c) {
    return __builtin_amdgcn_wmma_f32_16x16x32_f16(false, a, false, b,
                                                  (short)0, c, false, false);
}

// A-fragment (16x32 f16, documented CDNA5 layout) from a row-major [., ld] tile
__device__ __forceinline__ v16h load_a_frag_ld(const _Float16* As, int ld,
                                               int m0, int kc, int lane) {
    const _Float16* row = As + (m0 + (lane & 15)) * ld + kc;
    const int koff = (lane < 16) ? 0 : 8;
    v16h a;
#pragma unroll
    for (int v = 0; v < 8; ++v) {
        int k = (v < 4) ? (koff + 2 * v) : (16 + koff + 2 * (v - 4));
        a[2 * v]     = row[k];
        a[2 * v + 1] = row[k + 1];
    }
    return a;
}

#if HAVE_TDM
// 2D tile load global->LDS via Tensor Data Mover (D# groups built per ISA 08).
__device__ __forceinline__ void tdm_load_2d(unsigned lds_off, const void* gptr,
                                            unsigned td0, unsigned td1,
                                            unsigned tile0, unsigned tile1,
                                            unsigned stride0) {
    unsigned long long ga = (unsigned long long)(size_t)gptr;
    u32x4 g0;
    g0[0] = 1u;                                   // count=1, user mode
    g0[1] = lds_off;                              // lds_addr
    g0[2] = (unsigned)(ga & 0xffffffffull);       // global_addr lo
    g0[3] = (unsigned)((ga >> 32) & 0x01ffffffull) | (2u << 30); // hi | type=2
    i32x8 g1;
    g1[0] = (int)(1u << 16);                      // data_size=1 (2 bytes/elem)
    g1[1] = (int)((td0 & 0xffffu) << 16);         // tensor_dim0 lo16
    g1[2] = (int)(((td0 >> 16) & 0xffffu) | ((td1 & 0xffffu) << 16));
    g1[3] = (int)(((td1 >> 16) & 0xffffu) | ((tile0 & 0xffffu) << 16));
    g1[4] = (int)(tile1 & 0xffffu);               // tile_dim1, tile_dim2=0
    g1[5] = (int)stride0;                         // tensor_dim0_stride lo32
    g1[6] = 0;
    g1[7] = 0;
    i32x4 gz; gz[0] = gz[1] = gz[2] = gz[3] = 0;
#if defined(__clang_major__) && (__clang_major__ >= 23)
    i32x8 gz8;
#pragma unroll
    for (int q = 0; q < 8; ++q) gz8[q] = 0;
    __builtin_amdgcn_tensor_load_to_lds(g0, g1, gz, gz, gz8, 0);
#else
    __builtin_amdgcn_tensor_load_to_lds(g0, g1, gz, gz, 0);
#endif
}
#endif

// ------------------------------------------------------------------------
// k0: precompute iRFFT basis WMMA A-fragments (Cr and Ci), f16.
// ------------------------------------------------------------------------
__global__ void k0_afrag(_Float16* __restrict__ arT, _Float16* __restrict__ aiT) {
    const int t = threadIdx.x;               // 256 threads = 2 kk * 4 ti * 32 lanes
    const int lane = t & 31, ti = (t >> 5) & 3, kkI = (t >> 7) & 1;
    const int kk  = kkI * 32;
    const int tau = ti * 16 + (lane & 15);
    const int koff = (lane < 16) ? 0 : 8;
    _Float16 ar[16], ai[16];
#pragma unroll
    for (int v = 0; v < 8; ++v) {
        int kb = (v < 4) ? (koff + 2 * v) : (16 + koff + 2 * (v - 4));
#pragma unroll
        for (int h = 0; h < 2; ++h) {
            int f = kk + kb + h;
            float cf = (f > 32) ? 0.f
                     : ((f == 0 || f == 32) ? (1.f / 64.f) : (2.f / 64.f));
            float sv, cv;
            __sincosf(TWO_PI_64 * (float)(f * tau), &sv, &cv);
            ar[2 * v + h] = (_Float16)(cf * cv);
            ai[2 * v + h] = (_Float16)(-cf * sv);
        }
    }
    _Float16* dr = arT + ((size_t)(kkI * 4 + ti) * 32 + lane) * 16;
    _Float16* di = aiT + ((size_t)(kkI * 4 + ti) * 32 + lane) * 16;
#pragma unroll
    for (int q = 0; q < 16; ++q) { dr[q] = ar[q]; di[q] = ai[q]; }
}

// ------------------------------------------------------------------------
// k1: collapse weights against embeddings: vW[a][w][f] = sum_e emb[e]*Wa[w,e,f]
// ------------------------------------------------------------------------
__global__ void k1_collapse(const float* __restrict__ Wr,  const float* __restrict__ Wi,
                            const float* __restrict__ Wrl, const float* __restrict__ Wil,
                            const float* __restrict__ Wrr, const float* __restrict__ Wir,
                            const float* __restrict__ emb, float* __restrict__ vW) {
    const int w = blockIdx.x, a = blockIdx.y, f = threadIdx.x;
    const float* src = Wr;
    switch (a) {
        case 0: src = Wr;  break; case 1: src = Wi;  break;
        case 2: src = Wrl; break; case 3: src = Wil; break;
        case 4: src = Wrr; break; case 5: src = Wir; break;
    }
    const float* base = src + (size_t)w * E_ * E_;
    float s = 0.f;
    for (int e = 0; e < E_; ++e) s += emb[e] * base[e * E_ + f];
    vW[((size_t)a * W_ + w) * E_ + f] = s;
}

// ------------------------------------------------------------------------
// k2: per-(b,n,w) windowed 64-pt DFT (33 bins), magnitude ranking (top-32).
// ------------------------------------------------------------------------
__global__ void k2_stft(const float* __restrict__ x, float* __restrict__ SmRe,
                        float* __restrict__ SmIm, int* __restrict__ binRk) {
    __shared__ float fr[NFFT];
    __shared__ float Sre[F_], Sim[F_], mag[F_];
    const int blk = blockIdx.x;                 // = ((b*32+n)*17 + w)
    const int w   = blk % W_;
    const int bn  = blk / W_;
    const int n   = bn % NCH, b = bn / NCH;
    const int j   = threadIdx.x;
    int s = w * HOP + j - 32;                   // reflect pad by 32
    if (s < 0) s = -s;
    if (s > T_ - 1) s = 2 * (T_ - 1) - s;
    fr[j] = x[((size_t)b * T_ + s) * NCH + n] * hannw(j);
    __syncthreads();
    if (j < F_) {
        float re = 0.f, im = 0.f;
        for (int tt = 0; tt < NFFT; ++tt) {
            float sv, cv;
            __sincosf(TWO_PI_64 * (float)(j * tt), &sv, &cv);
            re += fr[tt] * cv;
            im -= fr[tt] * sv;
        }
        Sre[j] = re; Sim[j] = im; mag[j] = re * re + im * im;
    }
    __syncthreads();
    if (j < F_) {
        const float mj = mag[j];
        int rank = 0;
        for (int g = 0; g < F_; ++g) {
            float mg = mag[g];
            rank += (mg > mj) || (mg == mj && g < j);   // stable descending
        }
        if (rank < TOPM) {
            size_t base = (size_t)blk * TOPM;
            SmRe[base + rank] = Sre[j];
            SmIm[base + rank] = Sim[j];
            binRk[base + rank] = j;
        }
    }
}

// ------------------------------------------------------------------------
// k3: per-(b,n,w): rank-1 mixing + relu + softshrink -> Y[bin][e'] in LDS,
//     iRFFT via WMMA f16, hann window, env-normalized OLA.
//     wOff=0 -> even frames (tile [0,576) exactly: store)
//     wOff=1 -> odd frames (disjoint among themselves: plain accumulate)
// ------------------------------------------------------------------------
__global__ __launch_bounds__(256) void k3_synth(
        const float* __restrict__ vW,  const float* __restrict__ br,
        const float* __restrict__ bi,  const float* __restrict__ SmRe,
        const float* __restrict__ SmIm, const int* __restrict__ binRk,
        const _Float16* __restrict__ arT, const _Float16* __restrict__ aiT,
        float* __restrict__ ola, int wOff) {
    __shared__ _Float16 Yr[NFFT * E_];
    __shared__ _Float16 Yi[NFFT * E_];
    const int bn = blockIdx.x;
    const int w  = 2 * blockIdx.y + wOff;
    const int t  = threadIdx.x;

    unsigned* zr = (unsigned*)Yr; unsigned* zi = (unsigned*)Yi;
    for (int i = t; i < NFFT * E_ / 2; i += 256) { zr[i] = 0u; zi[i] = 0u; }
    __syncthreads();

    const float* vWr  = vW + ((size_t)0 * W_ + w) * E_;
    const float* vWi  = vW + ((size_t)1 * W_ + w) * E_;
    const float* vWrl = vW + ((size_t)2 * W_ + w) * E_;
    const float* vWil = vW + ((size_t)3 * W_ + w) * E_;
    const float* vWrr = vW + ((size_t)4 * W_ + w) * E_;
    const float* vWir = vW + ((size_t)5 * W_ + w) * E_;
    const size_t base = ((size_t)bn * W_ + w) * TOPM;

    for (int it = 0; it < 16; ++it) {
        int idx = t + 256 * it;                 // 0..4095 = (m,e)
        int m = idx >> 7, e = idx & 127;
        float are = SmRe[base + m], aim = SmIm[base + m];
        float lre = 0.f, lim = 0.f, rre = 0.f, rim = 0.f;
        if (w > 0)      { lre = SmRe[base - TOPM + m]; lim = SmIm[base - TOPM + m]; }
        if (w < W_ - 1) { rre = SmRe[base + TOPM + m]; rim = SmIm[base + TOPM + m]; }
        float c0 = vWr[e], c1 = vWi[e], c2 = vWrl[e], c3 = vWil[e],
              c4 = vWrr[e], c5 = vWir[e];
        float re = are*c0 - aim*c1 + lre*c2 - lim*c3 + rre*c4 - rim*c5 + br[w*E_ + e];
        float im = aim*c0 + are*c1 + lim*c2 + lre*c3 + rim*c4 + rre*c5 + bi[w*E_ + e];
        re = fmaxf(fmaxf(re, 0.f) - 0.01f, 0.f);   // relu then softshrink(>=0)
        im = fmaxf(fmaxf(im, 0.f) - 0.01f, 0.f);
        int bin = binRk[base + m];
        Yr[bin * E_ + e] = (_Float16)re;
        Yi[bin * E_ + e] = (_Float16)im;
    }
    __syncthreads();

    const int lane = t & 31, wv = t >> 5;
    const int n0 = wv * 16;                     // 8 waves x 16 e-cols
    v8f acc[4];
#pragma unroll
    for (int i = 0; i < 4; ++i)
#pragma unroll
        for (int q = 0; q < 8; ++q) acc[i][q] = 0.f;

#pragma unroll
    for (int kkI = 0; kkI < 2; ++kkI) {
        int kk = kkI * 32;
        v16h byr, byi;
#pragma unroll
        for (int s = 0; s < 16; ++s) {
            int k = kk + ((s < 8) ? (8 * (lane >> 4) + s) : (16 + 8 * (lane >> 4) + s - 8));
            byr[s] = Yr[k * E_ + n0 + (lane & 15)];
            byi[s] = Yi[k * E_ + n0 + (lane & 15)];
        }
#pragma unroll
        for (int ti = 0; ti < 4; ++ti) {
            v16h fa = *(const v16h*)(arT + ((size_t)(kkI * 4 + ti) * 32 + lane) * 16);
            v16h fb = *(const v16h*)(aiT + ((size_t)(kkI * 4 + ti) * 32 + lane) * 16);
            acc[ti] = wmma16(fa, byr, acc[ti]);
            acc[ti] = wmma16(fb, byi, acc[ti]);
        }
    }

    const int ecol = n0 + (lane & 15);
    const int rhi  = 8 * (lane >> 4);
#pragma unroll
    for (int ti = 0; ti < 4; ++ti) {
#pragma unroll
        for (int r = 0; r < 8; ++r) {
            int tau  = ti * 16 + r + rhi;
            int tpos = w * HOP + tau;
            float hwin = hannw(tau);
            float env = 0.f;
            int w1 = tpos >> 5, j1 = tpos & 31;
            if (w1 < W_) { float h1 = hannw(j1);      env += h1 * h1; }
            int w0 = w1 - 1;
            if (w0 >= 0) { float h0 = hannw(j1 + 32); env += h0 * h0; }
            float inv = (env > 1e-11f) ? 1.f / env : 1.f;
            float val = acc[ti][r] * hwin * inv;
            float* dst = ola + ((size_t)bn * L_ + tpos) * E_ + ecol;
            if (wOff) *dst += val; else *dst = val;
        }
    }
}

// ------------------------------------------------------------------------
// k4: crop OLA, add residual bias x[b,t,n]*emb[e], emit f16 GEMM A-matrix.
// ------------------------------------------------------------------------
__global__ void k4_xo(const float* __restrict__ ola, const float* __restrict__ x,
                      const float* __restrict__ emb, _Float16* __restrict__ xo) {
    size_t i = (size_t)blockIdx.x * 256 + threadIdx.x;   // < 512*65536
    int e = (int)(i & 127);
    size_t r = i >> 7;
    int t = (int)(r & 511);
    int bn = (int)(r >> 9);
    int b = bn >> 5, n = bn & 31;
    float v = ola[((size_t)bn * L_ + (t + 32)) * E_ + e]
            + x[((size_t)b * T_ + t) * NCH + n] * emb[e];
    xo[i] = (_Float16)v;
}

// k5: fc_w1 fp32 -> f16 (same 65536x512 layout), 4 elems/thread
__global__ void k5_cvt(const float* __restrict__ src, _Float16* __restrict__ dst) {
    size_t i = (size_t)blockIdx.x * 256 + threadIdx.x;
    const float4 v = ((const float4*)src)[i];
    _Float16 h4[4] = {(_Float16)v.x, (_Float16)v.y, (_Float16)v.z, (_Float16)v.w};
    *(uint2*)(dst + i * 4) = *(const uint2*)h4;
}

// ------------------------------------------------------------------------
// k6: fc1 GEMM  H(512x512) = leaky_relu( xo(512x65536) @ w1(65536x512) + b1 )
//     128x128 tile / block, 8 waves, K-stage 64, WMMA f16.
//     TDM path: double-buffered LDS, async tensor loads pipelined against
//     WMMA with s_wait_tensorcnt(2) (wait only for the previous stage).
// ------------------------------------------------------------------------
__global__ __launch_bounds__(256) void k6_fc1(const _Float16* __restrict__ A,
                                              const _Float16* __restrict__ Bw,
                                              const float* __restrict__ b1,
                                              float* __restrict__ Hout) {
    __shared__ _Float16 As[2][128 * KSTEP];   // [buf][m][k]   16 KB each
    __shared__ _Float16 Bs[2][KSTEP * 128];   // [buf][k][n]   16 KB each
    const int rowBase = blockIdx.y * 128, colBase = blockIdx.x * 128;
    const int t = threadIdx.x, lane = t & 31, wv = t >> 5;
    const int mBase = (wv >> 2) * 64;   // 2 wave-rows x 4 wave-cols
    const int nBase = (wv & 3) * 32;
    v8f acc[4][2];
#pragma unroll
    for (int i = 0; i < 4; ++i)
#pragma unroll
        for (int j = 0; j < 2; ++j)
#pragma unroll
            for (int q = 0; q < 8; ++q) acc[i][j][q] = 0.f;

    const int NST = KTOT / KSTEP;   // 1024 stages

#if HAVE_TDM
    if (wv == 0) {   // prologue: stage 0 tiles
        tdm_load_2d((unsigned)(size_t)(void*)As[0],
                    A + (size_t)rowBase * KTOT, KTOT, MBN, KSTEP, 128u, KTOT);
        tdm_load_2d((unsigned)(size_t)(void*)Bs[0],
                    Bw + colBase, HID, KTOT, 128u, KSTEP, HID);
    }
#pragma unroll 1
    for (int s = 0; s < NST; ++s) {
        const int buf = s & 1;
        if (wv == 0) {
            if (s + 1 < NST) {   // issue next stage, then wait for current
                int k0 = (s + 1) * KSTEP;
                tdm_load_2d((unsigned)(size_t)(void*)As[buf ^ 1],
                            A + (size_t)rowBase * KTOT + k0,
                            KTOT, MBN, KSTEP, 128u, KTOT);
                tdm_load_2d((unsigned)(size_t)(void*)Bs[buf ^ 1],
                            Bw + (size_t)k0 * HID + colBase,
                            HID, KTOT, 128u, KSTEP, HID);
                __builtin_amdgcn_s_wait_tensorcnt(2);
            } else {
                __builtin_amdgcn_s_wait_tensorcnt(0);
            }
        }
        __syncthreads();
        const _Float16* Ab = As[buf];
        const _Float16* Bb = Bs[buf];
#pragma unroll
        for (int kc = 0; kc < KSTEP; kc += 32) {
            v16h bf0, bf1;
#pragma unroll
            for (int s = 0; s < 16; ++s) {
                int k = kc + ((s < 8) ? (8 * (lane >> 4) + s) : (16 + 8 * (lane >> 4) + s - 8));
                bf0[s] = Bb[k * 128 + nBase + (lane & 15)];
                bf1[s] = Bb[k * 128 + nBase + 16 + (lane & 15)];
            }
#pragma unroll
            for (int i = 0; i < 4; ++i) {
                v16h af = load_a_frag_ld(Ab, KSTEP, mBase + 16 * i, kc, lane);
                acc[i][0] = wmma16(af, bf0, acc[i][0]);
                acc[i][1] = wmma16(af, bf1, acc[i][1]);
            }
        }
        __syncthreads();
    }
#else
#pragma unroll 1
    for (int s = 0; s < NST; ++s) {
        const int k0 = s * KSTEP;
        {   // cooperative tile loads (single buffer)
            int r = t >> 1, c = (t & 1) * 32;                 // A: 128 x 64 halves
            const uint4* sa = (const uint4*)(A + (size_t)(rowBase + r) * KTOT + k0 + c);
            uint4* da = (uint4*)(&As[0][r * KSTEP + c]);
            da[0] = sa[0]; da[1] = sa[1]; da[2] = sa[2]; da[3] = sa[3];
            int rb = t >> 2, cb = (t & 3) * 32;               // B: 64 x 128 halves
            const uint4* sb = (const uint4*)(Bw + (size_t)(k0 + rb) * HID + colBase + cb);
            uint4* db = (uint4*)(&Bs[0][rb * 128 + cb]);
            db[0] = sb[0]; db[1] = sb[1]; db[2] = sb[2]; db[3] = sb[3];
        }
        __syncthreads();
#pragma unroll
        for (int kc = 0; kc < KSTEP; kc += 32) {
            v16h bf0, bf1;
#pragma unroll
            for (int s = 0; s < 16; ++s) {
                int k = kc + ((s < 8) ? (8 * (lane >> 4) + s) : (16 + 8 * (lane >> 4) + s - 8));
                bf0[s] = Bs[0][k * 128 + nBase + (lane & 15)];
                bf1[s] = Bs[0][k * 128 + nBase + 16 + (lane & 15)];
            }
#pragma unroll
            for (int i = 0; i < 4; ++i) {
                v16h af = load_a_frag_ld(&As[0][0], KSTEP, mBase + 16 * i, kc, lane);
                acc[i][0] = wmma16(af, bf0, acc[i][0]);
                acc[i][1] = wmma16(af, bf1, acc[i][1]);
            }
        }
        __syncthreads();
    }
#endif

#pragma unroll
    for (int i = 0; i < 4; ++i)
#pragma unroll
        for (int j = 0; j < 2; ++j)
#pragma unroll
            for (int r = 0; r < 8; ++r) {
                int row = rowBase + mBase + 16 * i + r + 8 * (lane >> 4);
                int col = colBase + nBase + 16 * j + (lane & 15);
                float v = acc[i][j][r] + b1[col];
                v = (v > 0.f) ? v : 0.01f * v;                 // leaky_relu 0.01
                Hout[(size_t)row * HID + col] = v;
            }
}

// k7: fc2 + bias + output transpose (B,N,P) -> (B,P,N)
__global__ void k7_fc2(const float* __restrict__ H, const float* __restrict__ w2,
                       const float* __restrict__ b2, float* __restrict__ out) {
    int i = blockIdx.x * 256 + threadIdx.x;    // 512*96
    if (i >= MBN * PRED) return;
    int p = i % PRED, bn = i / PRED;
    float s = b2[p];
    for (int h = 0; h < HID; ++h) s += H[(size_t)bn * HID + h] * w2[(size_t)h * PRED + p];
    int b = bn >> 5, n = bn & 31;
    out[((size_t)b * PRED + p) * NCH + n] = s;
}

// ------------------------------------------------------------------------
extern "C" void kernel_launch(void* const* d_in, const int* in_sizes, int n_in,
                              void* d_out, int out_size, void* d_ws, size_t ws_size,
                              hipStream_t stream) {
    (void)in_sizes; (void)n_in; (void)out_size; (void)ws_size;
    const float* x   = (const float*)d_in[0];
    const float* emb = (const float*)d_in[1];
    const float* Wr  = (const float*)d_in[2];
    const float* Wi  = (const float*)d_in[3];
    const float* Wrl = (const float*)d_in[4];
    const float* Wil = (const float*)d_in[5];
    const float* Wrr = (const float*)d_in[6];
    const float* Wir = (const float*)d_in[7];
    const float* br  = (const float*)d_in[8];
    const float* bi  = (const float*)d_in[9];
    const float* w1  = (const float*)d_in[10];
    const float* b1  = (const float*)d_in[11];
    const float* w2  = (const float*)d_in[12];
    const float* b2  = (const float*)d_in[13];

    char* ws = (char*)d_ws;
    size_t off = 0;
    auto alloc = [&](size_t bytes) -> void* {
        void* p = ws + off;
        off += (bytes + 255) & ~(size_t)255;
        return p;
    };
    float*    vW    = (float*)alloc((size_t)6 * W_ * E_ * 4);
    float*    SmRe  = (float*)alloc((size_t)MBN * W_ * TOPM * 4);
    float*    SmIm  = (float*)alloc((size_t)MBN * W_ * TOPM * 4);
    int*      binRk = (int*)  alloc((size_t)MBN * W_ * TOPM * 4);
    _Float16* arT   = (_Float16*)alloc((size_t)8 * 32 * 16 * 2);
    _Float16* aiT   = (_Float16*)alloc((size_t)8 * 32 * 16 * 2);
    float*    ola   = (float*)alloc((size_t)MBN * L_ * E_ * 4);     // ~151 MB
    _Float16* xo    = (_Float16*)alloc((size_t)MBN * KTOT * 2);     //  64 MB
    _Float16* w1h   = (_Float16*)alloc((size_t)KTOT * HID * 2);     //  64 MB
    float*    hmid  = (float*)alloc((size_t)MBN * HID * 4);

    k0_afrag   <<<1, 256, 0, stream>>>(arT, aiT);
    k1_collapse<<<dim3(W_, 6), E_, 0, stream>>>(Wr, Wi, Wrl, Wil, Wrr, Wir, emb, vW);
    k2_stft    <<<B_ * NCH * W_, NFFT, 0, stream>>>(x, SmRe, SmIm, binRk);
    k3_synth   <<<dim3(MBN, 9), 256, 0, stream>>>(vW, br, bi, SmRe, SmIm, binRk,
                                                  arT, aiT, ola, 0);  // even frames
    k3_synth   <<<dim3(MBN, 8), 256, 0, stream>>>(vW, br, bi, SmRe, SmIm, binRk,
                                                  arT, aiT, ola, 1);  // odd frames
    k4_xo      <<<(MBN * (size_t)KTOT) / 256, 256, 0, stream>>>(ola, x, emb, xo);
    k5_cvt     <<<((size_t)KTOT * HID / 4) / 256, 256, 0, stream>>>(w1, w1h);
    k6_fc1     <<<dim3(4, 4), 256, 0, stream>>>(xo, w1h, b1, hmid);
    k7_fc2     <<<(MBN * PRED + 255) / 256, 256, 0, stream>>>(hmid, w2, b2, (float*)d_out);
}
